// DGLGCN_5634997092536
// MI455X (gfx1250) — matrix operand order repaired and checked by reading, compiled, not run on hardware
//
#include <hip/hip_runtime.h>

// GraphSAGE 2-layer forward for MI455X (gfx1250, wave32).
// - Aggregation: edge-parallel f32 atomic scatter-add (L2-resident: feat+agg = 102MB << 192MB L2).
// - GEMMs: v_wmma_f32_16x16x32_bf16; A tile (16x256 bf16, 8KB) staged into LDS once per block via
//   GLOBAL_LOAD_ASYNC_TO_LDS_B128 (+ s_wait_asynccnt) when available, ds_load_b128 fragment reads.
// - concat + mean-divide + bf16 convert done once per layer in a pack pass (no per-wave redundancy).

typedef __attribute__((ext_vector_type(16))) __bf16 v16bf;
typedef __attribute__((ext_vector_type(8)))  __bf16 v8bf;
typedef __attribute__((ext_vector_type(8)))  float  v8f;

#define D_IN  128
#define D_HID 128
#define D_OUT 64
#define D_CAT 256

#if defined(__has_builtin)
#if __has_builtin(__builtin_amdgcn_global_load_async_to_lds_b128)
#define HAVE_ASYNC_LDS 1
#endif
#endif

// builtin parameter types per hipcc diagnostic: pointer to 16B int vector, AS1 (global) / AS3 (LDS)
typedef int v4i __attribute__((vector_size(16)));
typedef __attribute__((address_space(1))) v4i gv4i;
typedef __attribute__((address_space(3))) v4i lv4i;

__device__ __forceinline__ unsigned short f2bfu(float f) {
  union { float f; unsigned u; } v; v.f = f;
  unsigned r = v.u + 0x7FFFu + ((v.u >> 16) & 1u);   // round-to-nearest-even
  return (unsigned short)(r >> 16);
}
__device__ __forceinline__ float bf2f(unsigned short h) {
  union { unsigned u; float f; } o; o.u = ((unsigned)h) << 16;
  return o.f;
}

// ---------------- utility kernels ----------------

__global__ void k_zero(float* __restrict__ p, long long n) {
  long long i = (long long)blockIdx.x * blockDim.x + threadIdx.x;
  long long s = (long long)gridDim.x * blockDim.x;
  for (; i < n; i += s) p[i] = 0.0f;
}

__global__ void k_cvt_w(const float* __restrict__ w, unsigned short* __restrict__ o, int n) {
  int i = blockIdx.x * blockDim.x + threadIdx.x;
  if (i < n) o[i] = f2bfu(w[i]);
}

__global__ void k_degree(const int* __restrict__ dst, int* __restrict__ cnt, int n_edges) {
  int e = blockIdx.x * blockDim.x + threadIdx.x;
  if (e < n_edges) atomicAdd(&cnt[dst[e]], 1);
}

__global__ void k_inv(const int* __restrict__ cnt, float* __restrict__ inv, int n) {
  int v = blockIdx.x * blockDim.x + threadIdx.x;
  if (v < n) inv[v] = 1.0f / fmaxf((float)cnt[v], 1.0f);
}

// ---------------- edge aggregation (scatter-add); one wave per edge ----------------

__global__ void k_agg_f32(const float* __restrict__ feat, const int* __restrict__ src,
                          const int* __restrict__ dst, float* __restrict__ agg, int n_edges) {
  int e    = blockIdx.x * (blockDim.x >> 5) + (threadIdx.x >> 5);
  int lane = threadIdx.x & 31;
  if (e >= n_edges) return;
  int s = src[e], d = dst[e];
  float4 v = ((const float4*)(feat + (size_t)s * D_IN))[lane];
  float* o = agg + (size_t)d * D_IN + lane * 4;
  atomicAdd(o + 0, v.x); atomicAdd(o + 1, v.y);
  atomicAdd(o + 2, v.z); atomicAdd(o + 3, v.w);
}

// h rows live in X with row-stride `stride` (bf16); aggregate cols 0..127 in f32.
__global__ void k_agg_bf16(const unsigned short* __restrict__ h, const int* __restrict__ src,
                           const int* __restrict__ dst, float* __restrict__ agg,
                           int n_edges, int stride) {
  int e    = blockIdx.x * (blockDim.x >> 5) + (threadIdx.x >> 5);
  int lane = threadIdx.x & 31;
  if (e >= n_edges) return;
  int s = src[e], d = dst[e];
  ushort4 v = ((const ushort4*)(h + (size_t)s * stride))[lane];
  float* o = agg + (size_t)d * D_HID + lane * 4;
  atomicAdd(o + 0, bf2f(v.x)); atomicAdd(o + 1, bf2f(v.y));
  atomicAdd(o + 2, bf2f(v.z)); atomicAdd(o + 3, bf2f(v.w));
}

// ---------------- pack kernels: build bf16 concat input X[v] = [u(v) | agg(v)*inv(v)] ----------------

__global__ void k_pack0(const float* __restrict__ feat, const float* __restrict__ agg,
                        const float* __restrict__ inv, unsigned short* __restrict__ X, int n) {
  int i = blockIdx.x * blockDim.x + threadIdx.x;         // over n*32 float4 groups
  if (i >= n * 32) return;
  int v = i >> 5, c4 = (i & 31) * 4;
  float4 f = *(const float4*)(feat + (size_t)v * D_IN + c4);
  float4 g = *(const float4*)(agg  + (size_t)v * D_IN + c4);
  float iv = inv[v];
  *(ushort4*)(X + (size_t)v * D_CAT + c4) =
      make_ushort4(f2bfu(f.x), f2bfu(f.y), f2bfu(f.z), f2bfu(f.w));
  *(ushort4*)(X + (size_t)v * D_CAT + D_IN + c4) =
      make_ushort4(f2bfu(g.x * iv), f2bfu(g.y * iv), f2bfu(g.z * iv), f2bfu(g.w * iv));
}

// second half only (layer 1: first half of X already holds h from in-place gemm0 store)
__global__ void k_pack1(const float* __restrict__ agg, const float* __restrict__ inv,
                        unsigned short* __restrict__ X, int n) {
  int i = blockIdx.x * blockDim.x + threadIdx.x;
  if (i >= n * 32) return;
  int v = i >> 5, c4 = (i & 31) * 4;
  float4 g = *(const float4*)(agg + (size_t)v * D_HID + c4);
  float iv = inv[v];
  *(ushort4*)(X + (size_t)v * D_CAT + D_HID + c4) =
      make_ushort4(f2bfu(g.x * iv), f2bfu(g.y * iv), f2bfu(g.z * iv), f2bfu(g.w * iv));
}

// ---------------- WMMA GEMM: Y = act(X @ W^T), X:[n,256] bf16, W:[NW*16,256] bf16 ----------------
// NW waves per block; wave w owns output cols [w*16, w*16+16); block owns 16 rows.
// A tile staged in LDS once per block; per k-step: 2x ds_load_b128 (A) + 1 global v16bf (B) + 1 WMMA.

template <int NW, bool RELU>
__global__ void k_gemm(const unsigned short* __restrict__ X,
                       const unsigned short* __restrict__ Wbf,
                       unsigned short* __restrict__ out_bf,   // RELU: bf16, row-stride D_CAT (in-place X)
                       float* __restrict__ out_f,             // !RELU: f32, row-stride D_OUT
                       int n) {
  __shared__ unsigned short smem[16 * D_CAT];                 // 8KB A tile
  int m_base = blockIdx.x * 16;
  if (m_base >= n) return;

  // ---- cooperative stage of A tile: 16 rows x 512B = 512 x 16B chunks ----
  for (int t = threadIdx.x; t < 512; t += NW * 32) {
    int row = t >> 5, chunk = t & 31;
    int grow = m_base + row; if (grow >= n) grow = n - 1;
    const unsigned short* g = X + (size_t)grow * D_CAT + chunk * 8;
    unsigned short* l = &smem[row * D_CAT + chunk * 8];
#ifdef HAVE_ASYNC_LDS
    __builtin_amdgcn_global_load_async_to_lds_b128((gv4i*)g, (lv4i*)l, 0, 0);
#else
    *(uint4*)l = *(const uint4*)g;
#endif
  }
#ifdef HAVE_ASYNC_LDS
#if __has_builtin(__builtin_amdgcn_s_wait_asynccnt)
  __builtin_amdgcn_s_wait_asynccnt(0);
#else
  asm volatile("s_wait_asynccnt 0x0" ::: "memory");
#endif
#endif
  __syncthreads();

  int wave = threadIdx.x >> 5;
  int lane = threadIdx.x & 31;
  int m  = lane & 15;            // A row within tile / B column within wave tile
  int hi = lane >> 4;            // K-half selector (A) / khalf (B)
  const unsigned short* arow = &smem[m * D_CAT];
  const unsigned short* wrow = Wbf + (size_t)(wave * 16 + m) * D_CAT + hi * 16;

  v8f c = {};
#pragma unroll
  for (int kb = 0; kb < 8; ++kb) {
    int k0 = kb * 32 + hi * 16;
    v8bf q0 = *(const v8bf*)(arow + k0);        // ds_load_b128: A elems 0..7  (K = kb*32+hi*16 ..)
    v8bf q1 = *(const v8bf*)(arow + k0 + 8);    // ds_load_b128: A elems 8..15 (K += 8)
    v16bf a = __builtin_shufflevector(q0, q1, 0, 1, 2, 3, 4, 5, 6, 7,
                                              8, 9, 10, 11, 12, 13, 14, 15);
    v16bf b = *(const v16bf*)(wrow + kb * 32);  // 16 consecutive K of W row (32B)
    c = __builtin_amdgcn_wmma_f32_16x16x32_bf16(false, a, false, b, (short)0, c, false, false);
  }

  // ---- store: col = wave*16 + (lane&15); rows m_base + hi*8 + r ----
  int col = wave * 16 + (lane & 15);
  int r0  = m_base + hi * 8;
#pragma unroll
  for (int r = 0; r < 8; ++r) {
    if (r0 + r < n) {
      if (RELU) {
        float v = c[r] > 0.0f ? c[r] : 0.0f;
        out_bf[(size_t)(r0 + r) * D_CAT + col] = f2bfu(v);
      } else {
        out_f[(size_t)(r0 + r) * D_OUT + col] = c[r];
      }
    }
  }
}

// ---------------- launcher ----------------

extern "C" void kernel_launch(void* const* d_in, const int* in_sizes, int n_in,
                              void* d_out, int out_size, void* d_ws, size_t ws_size,
                              hipStream_t stream) {
  (void)n_in; (void)out_size; (void)ws_size;
  const float* feat = (const float*)d_in[0];
  const int*   esrc = (const int*)d_in[1];
  const int*   edst = (const int*)d_in[2];
  const float* W0   = (const float*)d_in[3];
  const float* W1   = (const float*)d_in[4];
  float* out = (float*)d_out;

  int n       = in_sizes[0] / D_IN;     // 100000
  int n_edges = in_sizes[1];            // 1600000

  // workspace carve-up (chunks keep >=32B alignment)
  char* w = (char*)d_ws;
  float* agg = (float*)w;                        w += (size_t)n * D_IN * sizeof(float);
  int*   cnt = (int*)w;                          w += (size_t)n * sizeof(int);
  float* inv = (float*)w;                        w += (size_t)n * sizeof(float);
  unsigned short* X    = (unsigned short*)w;     w += (size_t)n * D_CAT * sizeof(unsigned short);
  unsigned short* w0bf = (unsigned short*)w;     w += (size_t)D_HID * D_CAT * sizeof(unsigned short);
  unsigned short* w1bf = (unsigned short*)w;

  int eb  = (n_edges + 255) / 256;
  int ewb = (n_edges + 7) / 8;          // 8 edge-waves per 256-thread block
  int mb  = (n + 15) / 16;
  int pb  = (n * 32 + 255) / 256;       // pack kernels: one float4 group per thread

  // init: zero agg+cnt (contiguous), weights -> bf16
  k_zero<<<4096, 256, 0, stream>>>(agg, (long long)n * D_IN + n);
  k_cvt_w<<<(D_HID * D_CAT + 255) / 256, 256, 0, stream>>>(W0, w0bf, D_HID * D_CAT);
  k_cvt_w<<<(D_OUT * D_CAT + 255) / 256, 256, 0, stream>>>(W1, w1bf, D_OUT * D_CAT);

  // layer 0
  k_degree<<<eb, 256, 0, stream>>>(edst, cnt, n_edges);
  k_inv<<<(n + 255) / 256, 256, 0, stream>>>(cnt, inv, n);
  k_agg_f32<<<ewb, 256, 0, stream>>>(feat, esrc, edst, agg, n_edges);
  k_pack0<<<pb, 256, 0, stream>>>(feat, agg, inv, X, n);
  // h = relu(X @ W0^T) written in place into X[:,0:128] (blocks own disjoint 16-row slices;
  // each block's global reads are complete -- asynccnt + barrier -- before its stores)
  k_gemm<8, true><<<mb, 256, 0, stream>>>(X, w0bf, X, nullptr, n);

  // layer 1 (reuse agg)
  k_zero<<<4096, 256, 0, stream>>>(agg, (long long)n * D_HID);
  k_agg_bf16<<<ewb, 256, 0, stream>>>(X, esrc, edst, agg, n_edges, D_CAT);
  k_pack1<<<pb, 256, 0, stream>>>(agg, inv, X, n);
  k_gemm<4, false><<<mb, 128, 0, stream>>>(X, w1bf, nullptr, out, n);
}
